// MultiHeadAttentionCross_25142738551386
// MI455X (gfx1250) — matrix operand order repaired and checked by reading, not compiled, not run on hardware
//
#include <hip/hip_runtime.h>
#include <hip/hip_bf16.h>

// MI455X / gfx1250, wave32. All matmuls via v_wmma_f32_16x16x32_bf16.
// Problem constants (from reference): B=16, C=512, Tt=2048, Ts=512, H=8, Dh=64.

typedef __attribute__((ext_vector_type(16))) __bf16        v16bf;
typedef __attribute__((ext_vector_type(8)))  float         v8f;
typedef __attribute__((ext_vector_type(4)))  float         v4f;
typedef __attribute__((ext_vector_type(4)))  unsigned int  v4u;
typedef __attribute__((ext_vector_type(2)))  unsigned int  v2u;

union BFrag { v16bf v; v4u q[2]; };

__device__ __forceinline__ unsigned short f2bf(float f) {
  unsigned u = __float_as_uint(f);
  u += 0x7FFFu + ((u >> 16) & 1u);   // round-to-nearest-even
  return (unsigned short)(u >> 16);
}

// A-matrix fragment (bf16 16x32, MxK): lane ln = M row, kh = lane>>4.
__device__ __forceinline__ v16bf frag_a(const unsigned short* row, int kb, int kh) {
  BFrag f;
  f.q[0] = *(const v4u*)(row + kb + kh * 8);
  f.q[1] = *(const v4u*)(row + kb + 16 + kh * 8);
  return f.v;
}
// B-matrix fragment (bf16 32x16, KxN): lane ln = N col; K in same order as frag_a from transposed storage.
__device__ __forceinline__ v16bf frag_b(const unsigned short* row, int kb, int kh) {
  BFrag f;
  f.q[0] = *(const v4u*)(row + kb + kh * 8);
  f.q[1] = *(const v4u*)(row + kb + 16 + kh * 8);
  return f.v;
}

#define WMMA_BF16(A, Bv, Cv) \
  __builtin_amdgcn_wmma_f32_16x16x32_bf16(false, (A), false, (Bv), (short)0, (Cv), false, false)

// Cooperative 16x512 bf16 (16KB) tile stage: 256 threads x 4 x b128.
__device__ __forceinline__ void stage16x512(const unsigned short* __restrict__ g,
                                            unsigned short* __restrict__ l) {
  const v4u* gs = (const v4u*)g;
  v4u* ls = (v4u*)l;
  int tid = threadIdx.x;
#pragma unroll
  for (int i = 0; i < 4; ++i) ls[tid + 256 * i] = gs[tid + 256 * i];
}

// Software-pipelined GEMM core: A (16x512) from LDS, four B row streams from memory.
// Computes acc[j] += A(16x512) x B_j(512x16) as 16 chained 16x16x32 WMMAs per j.
__device__ __forceinline__ void gemm_core_ldsA(const unsigned short* __restrict__ ldsA,
                                               const unsigned short* __restrict__ b0row,
                                               const unsigned short* __restrict__ b1row,
                                               const unsigned short* __restrict__ b2row,
                                               const unsigned short* __restrict__ b3row,
                                               int ln, int kh, v8f acc[4]) {
  const unsigned short* arow = ldsA + ln * 512;
  v16bf ac  = frag_a(arow, 0, kh);
  v16bf bc0 = frag_b(b0row, 0, kh);
  v16bf bc1 = frag_b(b1row, 0, kh);
  v16bf bc2 = frag_b(b2row, 0, kh);
  v16bf bc3 = frag_b(b3row, 0, kh);
  for (int kk = 0; kk < 15; ++kk) {
    int kb = kk * 32 + 32;
    v16bf an  = frag_a(arow, kb, kh);      // next-iter loads issue before this-iter WMMAs
    v16bf bn0 = frag_b(b0row, kb, kh);
    v16bf bn1 = frag_b(b1row, kb, kh);
    v16bf bn2 = frag_b(b2row, kb, kh);
    v16bf bn3 = frag_b(b3row, kb, kh);
    acc[0] = WMMA_BF16(ac, bc0, acc[0]);
    acc[1] = WMMA_BF16(ac, bc1, acc[1]);
    acc[2] = WMMA_BF16(ac, bc2, acc[2]);
    acc[3] = WMMA_BF16(ac, bc3, acc[3]);
    ac = an; bc0 = bn0; bc1 = bn1; bc2 = bn2; bc3 = bn3;
  }
  acc[0] = WMMA_BF16(ac, bc0, acc[0]);
  acc[1] = WMMA_BF16(ac, bc1, acc[1]);
  acc[2] = WMMA_BF16(ac, bc2, acc[2]);
  acc[3] = WMMA_BF16(ac, bc3, acc[3]);
}

// ---------------------------------------------------------------- pack weights
__global__ void pack_weights_kernel(const float* __restrict__ wq, const float* __restrict__ wk,
                                    const float* __restrict__ wv, const float* __restrict__ wo,
                                    unsigned short* __restrict__ dst) {
  int idx = blockIdx.x * blockDim.x + threadIdx.x;   // 0 .. 4*512*512-1
  int which = idx >> 18;
  int off = idx & 0x3FFFF;
  const float* s = (which == 0) ? wq : (which == 1) ? wk : (which == 2) ? wv : wo;
  dst[idx] = f2bf(s[off]);
}

// -------------------------------------------------- transpose (b,C,T)->(b,T,C) + bf16
__global__ void transpose_pack_kernel(const float* __restrict__ src, unsigned short* __restrict__ dst, int T) {
  __shared__ float tile[32][33];
  int b = blockIdx.z;
  int t0 = blockIdx.x * 32;
  int c0 = blockIdx.y * 32;
  const float* s = src + ((size_t)b * 512 + c0) * T + t0;
#pragma unroll
  for (int i = 0; i < 32; i += 8)
    tile[threadIdx.y + i][threadIdx.x] = s[(size_t)(threadIdx.y + i) * T + threadIdx.x];
  __syncthreads();
  unsigned short* d = dst + ((size_t)b * T + t0) * 512 + c0;
#pragma unroll
  for (int i = 0; i < 32; i += 8)
    d[(size_t)(threadIdx.y + i) * 512 + threadIdx.x] = f2bf(tile[threadIdx.x][threadIdx.y + i]);
}

// --------------------------------------- Q/K projection + fused RoPE (M=t, N=o)
// Block = 8 waves sharing one 16-row activation tile (staged in LDS); wave widx owns head widx.
__global__ __launch_bounds__(256) void proj_rope_kernel(const unsigned short* __restrict__ actT,
                                                        const unsigned short* __restrict__ W,
                                                        const float* __restrict__ bias,
                                                        unsigned short* __restrict__ outbf, int T) {
  __shared__ unsigned short s_a[16 * 512];   // 16KB shared A tile (rows t0..t0+15)
  int ttiles = T >> 4;
  int tb = blockIdx.x % ttiles;
  int b  = blockIdx.x / ttiles;
  int t0 = tb * 16;
  stage16x512(actT + ((size_t)b * T + t0) * 512, s_a);
  __syncthreads();

  int lane = threadIdx.x & 31;
  int ob = threadIdx.x >> 5;          // head / 64-wide o block
  int ln = lane & 15, kh = lane >> 4;
  int o0 = ob * 64;

  v8f acc[4] = {};
  gemm_core_ldsA(s_a,
                 W + (size_t)(o0 + ln) * 512,
                 W + (size_t)(o0 + 16 + ln) * 512,
                 W + (size_t)(o0 + 32 + ln) * 512,
                 W + (size_t)(o0 + 48 + ln) * 512,
                 ln, kh, acc);

  float bv0 = bias[o0 + ln],      bv1 = bias[o0 + 16 + ln];
  float bv2 = bias[o0 + 32 + ln], bv3 = bias[o0 + 48 + ln];
  // theta_j = 10000^{-j/16}; RoPE pairs (d, d+16) live in the same lane, adjacent accumulators.
  float theta = __powf(10000.f, -(float)ln * (1.f / 16.f));
  unsigned short* obase = outbf + ((size_t)(b * 8 + ob) * T) * 64;
#pragma unroll
  for (int r = 0; r < 8; ++r) {
    int t = t0 + r + 8 * kh;          // D-matrix row: M = r + 8*(lane>=16)
    float sn, cs;
    __sincosf((float)t * theta, &sn, &cs);
    float v0 = acc[0][r] + bv0;
    float v1 = acc[1][r] + bv1;
    unsigned short* orow = obase + (size_t)t * 64;
    orow[ln]      = f2bf(v0 * cs - v1 * sn);   // d in [0,16)
    orow[16 + ln] = f2bf(v1 * cs + v0 * sn);   // d in [16,32)
    orow[32 + ln] = f2bf(acc[2][r] + bv2);     // pass-through
    orow[48 + ln] = f2bf(acc[3][r] + bv3);
  }
}

// V projection: A = W rows (staged in LDS, shared by block), B = cT rows. -> vT (B,H,64,Ts) bf16
__global__ __launch_bounds__(256) void proj_v_kernel(const unsigned short* __restrict__ W,
                                                     const unsigned short* __restrict__ cT,
                                                     const float* __restrict__ bias,
                                                     unsigned short* __restrict__ vT) {
  __shared__ unsigned short s_w[16 * 512];
  int ot = blockIdx.x & 31;
  int b  = blockIdx.x >> 5;
  int o0 = ot * 16;
  stage16x512(W + (size_t)o0 * 512, s_w);
  __syncthreads();

  int lane = threadIdx.x & 31;
  int s0 = (threadIdx.x >> 5) * 64;   // wave owns 64 source columns
  int ln = lane & 15, kh = lane >> 4;
  const unsigned short* act = cT + (size_t)b * 512 * 512;
  v8f acc[4] = {};
  gemm_core_ldsA(s_w,
                 act + (size_t)(s0 + ln) * 512,
                 act + (size_t)(s0 + 16 + ln) * 512,
                 act + (size_t)(s0 + 32 + ln) * 512,
                 act + (size_t)(s0 + 48 + ln) * 512,
                 ln, kh, acc);
#pragma unroll
  for (int r = 0; r < 8; ++r) {
    int o = o0 + r + 8 * kh;
    float bv = bias[o];
    unsigned short* orow = vT + (((size_t)(b * 8 + (o >> 6)) * 64 + (o & 63)) * 512) + s0;
#pragma unroll
    for (int j = 0; j < 4; ++j) orow[j * 16 + ln] = f2bf(acc[j][r] + bv);
  }
}

// Output projection: A = Wo rows (LDS-staged), B = outT rows -> y[b,o,t] fp32 coalesced along t
__global__ __launch_bounds__(256) void proj_o_kernel(const unsigned short* __restrict__ W,
                                                     const unsigned short* __restrict__ outT,
                                                     const float* __restrict__ bias,
                                                     float* __restrict__ y) {
  __shared__ unsigned short s_w[16 * 512];
  int nbblk = blockIdx.x & 3;
  int ot = (blockIdx.x >> 2) & 31;
  int b  = blockIdx.x >> 7;
  int o0 = ot * 16;
  stage16x512(W + (size_t)o0 * 512, s_w);
  __syncthreads();

  int lane = threadIdx.x & 31;
  int t0 = (nbblk * 8 + (threadIdx.x >> 5)) * 64;
  int ln = lane & 15, kh = lane >> 4;
  const unsigned short* act = outT + (size_t)b * 2048 * 512;
  v8f acc[4] = {};
  gemm_core_ldsA(s_w,
                 act + (size_t)(t0 + ln) * 512,
                 act + (size_t)(t0 + 16 + ln) * 512,
                 act + (size_t)(t0 + 32 + ln) * 512,
                 act + (size_t)(t0 + 48 + ln) * 512,
                 ln, kh, acc);
#pragma unroll
  for (int r = 0; r < 8; ++r) {
    int o = o0 + r + 8 * kh;
    float bv = bias[o];
    float* yrow = y + ((size_t)b * 512 + o) * 2048 + t0;
#pragma unroll
    for (int j = 0; j < 4; ++j) yrow[j * 16 + ln] = acc[j][r] + bv;
  }
}

// ------------------------------------------------- fused attention: scores->softmax->attn.v
// One wave per block owns 16 target rows x 512 source cols.
__global__ __launch_bounds__(32) void attn_kernel(const unsigned short* __restrict__ qbf,
                                                  const unsigned short* __restrict__ kbf,
                                                  const unsigned short* __restrict__ vT,
                                                  float* __restrict__ attn_out,
                                                  unsigned short* __restrict__ outT) {
  __shared__ float          s_sc[16 * 512];   // 32 KB scores / exp values
  __shared__ unsigned short s_p [16 * 512];   // 16 KB bf16 probabilities
  __shared__ float          s_inv[16];
  int lane = threadIdx.x;
  int ln = lane & 15, kh = lane >> 4;
  int tb = blockIdx.x & 127;
  int h  = (blockIdx.x >> 7) & 7;
  int b  = blockIdx.x >> 10;
  int t0 = tb * 16;
  size_t bh = (size_t)(b * 8 + h);

  // Q fragments (K = d = 64 -> two 16x32 A-frags), held for the whole strip
  const unsigned short* qrow = qbf + (bh * 2048 + t0 + ln) * 64;
  v16bf qa0 = frag_a(qrow, 0, kh);
  v16bf qa1 = frag_a(qrow, 32, kh);

  // scores = q . k^T * 1/sqrt(64), software-pipelined over 32 source tiles
  const unsigned short* kbase = kbf + (bh * 512 + ln) * 64;
  v16bf kc0 = frag_b(kbase, 0, kh);
  v16bf kc1 = frag_b(kbase, 32, kh);
  for (int st = 0; st < 32; ++st) {
    v16bf kn0 = kc0, kn1 = kc1;
    if (st < 31) {
      const unsigned short* kr = kbase + (size_t)(st + 1) * 16 * 64;
      kn0 = frag_b(kr, 0, kh);
      kn1 = frag_b(kr, 32, kh);
    }
    v8f acc = {};
    acc = WMMA_BF16(qa0, kc0, acc);
    acc = WMMA_BF16(qa1, kc1, acc);
    int s0 = st * 16;
#pragma unroll
    for (int r = 0; r < 8; ++r)
      s_sc[(r + 8 * kh) * 512 + s0 + ln] = acc[r] * 0.125f;
    kc0 = kn0; kc1 = kn1;
  }
  __syncthreads();

  // softmax reduction: lane ln owns row ln, half kh owns 256 cols; merge halves via shfl_xor(16)
  {
    float* row = s_sc + ln * 512 + kh * 256;
    float mx = -3.4e38f;
#pragma unroll 8
    for (int i = 0; i < 64; ++i) {
      v4f v = *(const v4f*)(row + i * 4);
      mx = fmaxf(mx, fmaxf(fmaxf(v.x, v.y), fmaxf(v.z, v.w)));
    }
    mx = fmaxf(mx, __shfl_xor(mx, 16, 32));
    float sum = 0.f;
#pragma unroll 8
    for (int i = 0; i < 64; ++i) {
      v4f v = *(const v4f*)(row + i * 4);
      v4f e;
      e.x = __expf(v.x - mx); e.y = __expf(v.y - mx);
      e.z = __expf(v.z - mx); e.w = __expf(v.w - mx);
      *(v4f*)(row + i * 4) = e;
      sum += (e.x + e.y) + (e.z + e.w);
    }
    sum += __shfl_xor(sum, 16, 32);
    if (kh == 0) s_inv[ln] = 1.f / sum;
  }
  __syncthreads();

  // normalize: b128 fp32 attn stores (dominant HBM stream) + packed bf16 probs to LDS
  float* gb = attn_out + (bh * 2048 + t0) * 512;
  for (int r = 0; r < 16; ++r) {
    float iv = s_inv[r];
    const float* srow = s_sc + r * 512;
    float* grow = gb + (size_t)r * 512;
    unsigned short* prow = s_p + r * 512;
#pragma unroll
    for (int i = 0; i < 4; ++i) {
      int s4 = (i * 32 + lane) * 4;
      v4f v = *(const v4f*)(srow + s4);
      v4f p = v * iv;
      *(v4f*)(grow + s4) = p;
      v2u pk;
      pk.x = (unsigned)f2bf(p.x) | ((unsigned)f2bf(p.y) << 16);
      pk.y = (unsigned)f2bf(p.z) | ((unsigned)f2bf(p.w) << 16);
      *(v2u*)(prow + s4) = pk;
    }
  }
  __syncthreads();

  // out(16 x 64) = attn(16 x 512) . v(512 x 64): A from LDS, pipelined core
  v8f acc2[4] = {};
  gemm_core_ldsA(s_p,
                 vT + (bh * 64 + ln) * 512,
                 vT + (bh * 64 + 16 + ln) * 512,
                 vT + (bh * 64 + 32 + ln) * 512,
                 vT + (bh * 64 + 48 + ln) * 512,
                 ln, kh, acc2);
#pragma unroll
  for (int r = 0; r < 8; ++r) {
    int t = t0 + r + 8 * kh;
    unsigned short* orow = outT + ((size_t)b * 2048 + t) * 512 + h * 64;
#pragma unroll
    for (int j = 0; j < 4; ++j) orow[j * 16 + ln] = f2bf(acc2[j][r]);
  }
}

// ---------------------------------------------------------------------- launch
extern "C" void kernel_launch(void* const* d_in, const int* in_sizes, int n_in,
                              void* d_out, int out_size, void* d_ws, size_t ws_size,
                              hipStream_t stream) {
  (void)in_sizes; (void)n_in; (void)out_size; (void)ws_size;
  const float* x  = (const float*)d_in[0];
  const float* c  = (const float*)d_in[1];
  const float* Wq = (const float*)d_in[2];
  const float* bq = (const float*)d_in[3];
  const float* Wk = (const float*)d_in[4];
  const float* bk = (const float*)d_in[5];
  const float* Wv = (const float*)d_in[6];
  const float* bv = (const float*)d_in[7];
  const float* Wo = (const float*)d_in[8];
  const float* bo = (const float*)d_in[9];

  float* y    = (float*)d_out;                       // (16,512,2048) fp32
  float* attn = y + (size_t)16 * 512 * 2048;         // (16,8,2048,512) fp32

  // workspace layout, total ~122 MB
  char* ws = (char*)d_ws;
  unsigned short* wqb  = (unsigned short*)(ws);                         // 4x 512KB contiguous
  unsigned short* wkb  = wqb + (size_t)512 * 512;
  unsigned short* wvb  = wkb + (size_t)512 * 512;
  unsigned short* wob  = wvb + (size_t)512 * 512;
  unsigned short* xT   = wob + (size_t)512 * 512;                       // (16,2048,512) 32MB
  unsigned short* cT   = xT  + (size_t)16 * 2048 * 512;                 // (16,512,512)   8MB
  unsigned short* qbf  = cT  + (size_t)16 * 512 * 512;                  // (16,8,2048,64) 32MB
  unsigned short* kbf  = qbf + (size_t)16 * 8 * 2048 * 64;              // (16,8,512,64)   8MB
  unsigned short* vTb  = kbf + (size_t)16 * 8 * 512 * 64;               // (16,8,64,512)   8MB
  unsigned short* outT = vTb + (size_t)16 * 8 * 64 * 512;               // (16,2048,512)  32MB

  pack_weights_kernel<<<4096, 256, 0, stream>>>(Wq, Wk, Wv, Wo, wqb);
  transpose_pack_kernel<<<dim3(64, 16, 16), dim3(32, 8), 0, stream>>>(x, xT, 2048);
  transpose_pack_kernel<<<dim3(16, 16, 16), dim3(32, 8), 0, stream>>>(c, cT, 512);
  // Q,K projections (block = one t-tile, 8 waves = 8 heads, A tile staged in LDS)
  proj_rope_kernel<<<16 * 128, 256, 0, stream>>>(xT, wqb, bq, qbf, 2048);
  proj_rope_kernel<<<16 * 32,  256, 0, stream>>>(cT, wkb, bk, kbf, 512);
  // V projection (block = one 16-row W tile, 8 waves cover 8 source-column blocks)
  proj_v_kernel<<<16 * 32, 256, 0, stream>>>(wvb, cT, bv, vTb);
  // fused attention: 16*8*128 strips, 1 wave each
  attn_kernel<<<16384, 32, 0, stream>>>(qbf, kbf, vTb, attn, outT);
  // output projection (block = one W tile, 8 waves cover 8 of 32 t-blocks)
  proj_o_kernel<<<16 * 32 * 4, 256, 0, stream>>>(wob, outT, bo, y);
}
